// GraphVLAD_50560355009105
// MI455X (gfx1250) — compile-verified, matrix operand order checked by reading, not run on hardware
//
#include <hip/hip_runtime.h>
#include <hip/hip_bf16.h>
#include <math.h>

// ---------------- problem constants ----------------
#define GDIM   4096
#define HDIM   2048
#define NB     5
#define NNODE  512
#define SUB    8
#define M_TOT  4096          // NNODE * SUB
#define K_TOT  4096          // GDIM
#define D_TOT  32768         // SUB * GDIM

typedef __attribute__((ext_vector_type(16))) __bf16 v16bf;
typedef __attribute__((ext_vector_type(8)))  __bf16 v8bf;
typedef __attribute__((ext_vector_type(8)))  float  v8f;
typedef __attribute__((ext_vector_type(4)))  float  v4f;

// ---------------------------------------------------------------------------
// Phase 1a: x_src (f32, viewed [4096 x 4096]) -> bf16, vectorized 8/thread
// ---------------------------------------------------------------------------
__global__ void cvt_bf16_kernel(const float* __restrict__ src,
                                __bf16* __restrict__ dst) {
    int t = blockIdx.x * blockDim.x + threadIdx.x;   // 0 .. 4096*4096/8 - 1
    const v4f* p = (const v4f*)src + (size_t)t * 2;
    v4f a = p[0];
    v4f b = p[1];
    v8bf o;
    o[0] = (__bf16)a[0]; o[1] = (__bf16)a[1]; o[2] = (__bf16)a[2]; o[3] = (__bf16)a[3];
    o[4] = (__bf16)b[0]; o[5] = (__bf16)b[1]; o[6] = (__bf16)b[2]; o[7] = (__bf16)b[3];
    ((v8bf*)dst)[t] = o;
}

// ---------------------------------------------------------------------------
// Phase 1b: neighbor sum over NB=5 rows, f32 -> bf16.  A_nb[m][g], m=n*8+s.
// ---------------------------------------------------------------------------
__global__ void nb_agg_kernel(const float* __restrict__ x_nb,
                              __bf16* __restrict__ A_nb) {
    int t  = blockIdx.x * blockDim.x + threadIdx.x;  // 0 .. 4096*512-1
    int m  = t >> 9;                                 // row (512 chunks of 8 per row)
    int gc = (t & 511) << 3;                         // starting g of 8-wide chunk
    int n  = m >> 3;
    int s  = m & 7;
    float acc[8];
#pragma unroll
    for (int i = 0; i < 8; ++i) acc[i] = 0.0f;
#pragma unroll
    for (int j = 0; j < NB; ++j) {
        const float* p = x_nb + (size_t)(n * NB + j) * D_TOT + (size_t)s * GDIM + gc;
        v4f a = *(const v4f*)p;
        v4f b = *(const v4f*)(p + 4);
#pragma unroll
        for (int i = 0; i < 4; ++i) { acc[i] += a[i]; acc[4 + i] += b[i]; }
    }
    v8bf o;
#pragma unroll
    for (int i = 0; i < 8; ++i) o[i] = (__bf16)acc[i];
    ((v8bf*)A_nb)[t] = o;
}

// ---------------------------------------------------------------------------
// Phase 1c: weight transpose + convert: w[g][h] f32 -> wT[h][g] bf16
// LDS-tiled 32x32 so both global read and write are coalesced.
// ---------------------------------------------------------------------------
__global__ void wtrans_kernel(const float* __restrict__ w,
                              __bf16* __restrict__ wT) {
    __shared__ float tile[32][33];
    int h0 = blockIdx.x * 32;        // over HDIM
    int g0 = blockIdx.y * 32;        // over GDIM
    int tx = threadIdx.x;            // 0..31
    int ty = threadIdx.y;            // 0..7
#pragma unroll
    for (int i = 0; i < 32; i += 8)
        tile[ty + i][tx] = w[(size_t)(g0 + ty + i) * HDIM + h0 + tx];
    __syncthreads();
#pragma unroll
    for (int i = 0; i < 32; i += 8)
        wT[(size_t)(h0 + ty + i) * GDIM + g0 + tx] = (__bf16)tile[tx][ty + i];
}

// ---------------------------------------------------------------------------
// Fragment load helper: 16 bf16 for one 16x16x32 WMMA operand, per ISA
// 7.12.2 16-bit layout. p already points at row/col base + k0 + half*8;
// lo = K[k0+8h .. +8), hi = K[k0+16+8h .. +8)  -> two b128 loads.
// ---------------------------------------------------------------------------
__device__ __forceinline__ v16bf load_frag(const __bf16* __restrict__ p) {
    v8bf lo = *(const v8bf*)p;
    v8bf hi = *(const v8bf*)(p + 16);
    return __builtin_shufflevector(lo, hi,
        0,1,2,3,4,5,6,7,8,9,10,11,12,13,14,15);
}

// ---------------------------------------------------------------------------
// Phase 2: fused dual GEMM + GELU + residual.
// blockIdx.z selects {self, nb}. Block tile 128x128, 8 waves (wave32),
// each wave a 4x2 grid of v_wmma_f32_16x16x32_bf16 (64x32 per wave).
// K-loop is software-pipelined 2 deep (register double-buffer) and
// prefetches 4 K-steps ahead with global_prefetch.
// ---------------------------------------------------------------------------
__global__ __launch_bounds__(256)
void gemm_gelu_kernel(const __bf16* __restrict__ Aself,
                      const __bf16* __restrict__ Anb,
                      const __bf16* __restrict__ BTself,
                      const __bf16* __restrict__ BTnb,
                      const float*  __restrict__ x_src,
                      float* __restrict__ out) {
    const int z    = blockIdx.z;
    const __bf16* __restrict__ A  = z ? Anb   : Aself;
    const __bf16* __restrict__ BT = z ? BTnb  : BTself;

    const int lane = threadIdx.x & 31;
    const int wid  = threadIdx.x >> 5;       // 0..7
    const int wm   = wid >> 2;               // 0..1  (64-row slabs)
    const int wn   = wid & 3;                // 0..3  (32-col slabs)
    const int row0 = blockIdx.y * 128 + wm * 64;
    const int col0 = blockIdx.x * 128 + wn * 32;
    const int half = lane >> 4;
    const int r    = lane & 15;

    // Per-fragment-stream base pointers (advance by k within the K loop).
    const __bf16* aBase[4];
    const __bf16* bBase[2];
#pragma unroll
    for (int mt = 0; mt < 4; ++mt)
        aBase[mt] = A + (size_t)(row0 + mt * 16 + r) * K_TOT + half * 8;
#pragma unroll
    for (int nt = 0; nt < 2; ++nt)
        bBase[nt] = BT + (size_t)(col0 + nt * 16 + r) * K_TOT + half * 8;

    v8f c[4][2] = {};

    // 2-deep software pipeline over K (steps of 32, unrolled x2).
    v16bf a0[4], a1[4], b0[2], b1[2];
#pragma unroll
    for (int mt = 0; mt < 4; ++mt) a0[mt] = load_frag(aBase[mt]);
#pragma unroll
    for (int nt = 0; nt < 2; ++nt) b0[nt] = load_frag(bBase[nt]);

    for (int k0 = 0; k0 < K_TOT; k0 += 64) {
        // Issue loads for k0+32 into buffer 1 (overlap with buffer-0 WMMAs).
#pragma unroll
        for (int mt = 0; mt < 4; ++mt) a1[mt] = load_frag(aBase[mt] + k0 + 32);
#pragma unroll
        for (int nt = 0; nt < 2; ++nt) b1[nt] = load_frag(bBase[nt] + k0 + 32);

        // Prefetch 4 K-steps ahead (64B-aligned region covers lo+hi chunks).
        const int kp = (k0 + 128) & (K_TOT - 1);
#pragma unroll
        for (int mt = 0; mt < 4; ++mt) __builtin_prefetch(aBase[mt] + kp, 0, 0);
#pragma unroll
        for (int nt = 0; nt < 2; ++nt) __builtin_prefetch(bBase[nt] + kp, 0, 0);

        // Compute with buffer 0 (k0).
#pragma unroll
        for (int mt = 0; mt < 4; ++mt)
#pragma unroll
            for (int nt = 0; nt < 2; ++nt)
                c[mt][nt] = __builtin_amdgcn_wmma_f32_16x16x32_bf16(
                    false, a0[mt], false, b0[nt],
                    (short)0, c[mt][nt], false, false);

        // Refill buffer 0 with k0+64 (wraps harmlessly in-bounds on last iter;
        // that data is never consumed).
        const int kn = (k0 + 64) & (K_TOT - 1);
#pragma unroll
        for (int mt = 0; mt < 4; ++mt) a0[mt] = load_frag(aBase[mt] + kn);
#pragma unroll
        for (int nt = 0; nt < 2; ++nt) b0[nt] = load_frag(bBase[nt] + kn);

        // Compute with buffer 1 (k0+32).
#pragma unroll
        for (int mt = 0; mt < 4; ++mt)
#pragma unroll
            for (int nt = 0; nt < 2; ++nt)
                c[mt][nt] = __builtin_amdgcn_wmma_f32_16x16x32_bf16(
                    false, a1[mt], false, b1[nt],
                    (short)0, c[mt][nt], false, false);
    }

    // Epilogue: C/D layout (ISA 7.12.2): VGPR v, lanes 0-15 -> M=v, N=lane;
    // lanes 16-31 -> M=v+8, N=lane-16.  Exact-erf GELU + residual.
#pragma unroll
    for (int mt = 0; mt < 4; ++mt) {
#pragma unroll
        for (int nt = 0; nt < 2; ++nt) {
            const int hcol = col0 + nt * 16 + r;
#pragma unroll
            for (int v = 0; v < 8; ++v) {
                const int m = row0 + mt * 16 + v + half * 8;
                const int n = m >> 3;
                const int s = m & 7;
                const size_t idx = (size_t)n * D_TOT + (size_t)s * GDIM
                                 + (size_t)z * HDIM + hcol;
                const float x = c[mt][nt][v];
                const float g = 0.5f * x * (1.0f + erff(x * 0.70710678118654752f));
                out[idx] = g + x_src[idx];
            }
        }
    }
}

// ---------------------------------------------------------------------------
// Host launcher
// ---------------------------------------------------------------------------
extern "C" void kernel_launch(void* const* d_in, const int* in_sizes, int n_in,
                              void* d_out, int out_size, void* d_ws, size_t ws_size,
                              hipStream_t stream) {
    const float* x_src  = (const float*)d_in[0];   // [512, 32768]
    const float* x_nb   = (const float*)d_in[1];   // [2560, 32768]
    const float* w_self = (const float*)d_in[2];   // [4096, 2048]
    const float* w_nb   = (const float*)d_in[3];   // [4096, 2048]
    float* out = (float*)d_out;                    // [512, 32768]

    char* ws = (char*)d_ws;
    __bf16* Aself  = (__bf16*)(ws);                         // 32 MB
    __bf16* Anb    = (__bf16*)(ws + (size_t)33554432);      // 32 MB
    __bf16* BTself = (__bf16*)(ws + (size_t)67108864);      // 16 MB
    __bf16* BTnb   = (__bf16*)(ws + (size_t)83886080);      // 16 MB (total 96 MB)

    // Phase 1: converts / aggregation / weight transposes (independent, same stream)
    cvt_bf16_kernel<<<dim3(M_TOT * K_TOT / 8 / 256), dim3(256), 0, stream>>>(x_src, Aself);
    nb_agg_kernel  <<<dim3(M_TOT * K_TOT / 8 / 256), dim3(256), 0, stream>>>(x_nb, Anb);
    wtrans_kernel  <<<dim3(HDIM / 32, GDIM / 32), dim3(32, 8), 0, stream>>>(w_self, BTself);
    wtrans_kernel  <<<dim3(HDIM / 32, GDIM / 32), dim3(32, 8), 0, stream>>>(w_nb, BTnb);

    // Phase 2: fused dual-GEMM + GELU + residual
    gemm_gelu_kernel<<<dim3(HDIM / 128, M_TOT / 128, 2), dim3(256), 0, stream>>>(
        Aself, Anb, BTself, BTnb, x_src, out);
}
